// TrajectoryModel4_81724637708791
// MI455X (gfx1250) — matrix operand order repaired and checked by reading, not compiled, not run on hardware
//
#include <hip/hip_runtime.h>
#include <stdint.h>

// ---------------------------------------------------------------------------
// Types for WMMA fragments (gfx1250, wave32)
// ---------------------------------------------------------------------------
typedef __attribute__((ext_vector_type(16))) _Float16 v16h;
typedef __attribute__((ext_vector_type(8)))  _Float16 v8h;
typedef __attribute__((ext_vector_type(8)))  float    v8f;

__device__ __forceinline__ v16h pack2(v8h lo, v8h hi) {
  v16h a;
#pragma unroll
  for (int e = 0; e < 8; ++e) { a[e] = lo[e]; a[e + 8] = hi[e]; }
  return a;
}
// A fragment (16x32 f16): lane holds row M=lane&15; e<8 -> k=8*half+e, e>=8 -> k=16+8*half+(e-8)
__device__ __forceinline__ v16h frag_a(const _Float16* row, int half) {
  return pack2(*(const v8h*)(row + 8 * half), *(const v8h*)(row + 16 + 8 * half));
}
// B fragment (32x16 f16) loaded from B^T rows ([n][k]): e<8 -> k=8*half+e, e>=8 -> k=16+8*half+(e-8)
__device__ __forceinline__ v16h frag_b(const _Float16* row, int half) {
  return pack2(*(const v8h*)(row + 8 * half), *(const v8h*)(row + 16 + 8 * half));
}
__device__ __forceinline__ v8f wmma_f16(v16h a, v16h b, v8f c) {
  return __builtin_amdgcn_wmma_f32_16x16x32_f16(false, a, false, b, (short)0, c, false, false);
}

// Async global -> LDS 16-byte copy (CDNA5, ASYNCcnt-tracked, no VGPR round-trip).
// LDS operand is the wave-relative LDS byte offset (low 32 bits of generic ptr).
__device__ __forceinline__ void async_cp16(const _Float16* lds_dst, const _Float16* gsrc) {
  uint32_t l = (uint32_t)(uintptr_t)lds_dst;
  asm volatile("global_load_async_to_lds_b128 %0, %1, off"
               :: "v"(l), "v"(gsrc) : "memory");
}

// ---------------------------------------------------------------------------
// GEMM: C[M,N] = act(A[M,K] @ W[K,N] + bias), A f16 [M,K], Wt f16 pre-transposed [N,K].
// Optional fp32 and/or f16 outputs. Block 256 thr = 8 waves (4x2), tile 128x64,
// wave tile 32x32 (2x2 WMMA). K-loop double-buffered via async LDS copies:
// each wave issues exactly 3 async b128 instructions per tile (2 for A, 1 for B).
// Requires M%128==0, N%64==0, K%32==0.
// ---------------------------------------------------------------------------
__global__ __launch_bounds__(256)
void gemm_kernel(const _Float16* __restrict__ A, const _Float16* __restrict__ Wt,
                 const float* __restrict__ bias, float* __restrict__ Cf,
                 _Float16* __restrict__ Ch, int M, int N, int K, int relu)
{
  __shared__ __align__(16) _Float16 sA[2][128][32];   // 2 x 8 KB
  __shared__ __align__(16) _Float16 sB[2][64][32];    // 2 x 4 KB
  const int tid  = threadIdx.x;
  const int wave = tid >> 5, lane = tid & 31;
  const int half = lane >> 4, l16 = lane & 15;
  const int wm = wave >> 1, wn = wave & 1;
  const int m_blk = blockIdx.x * 128, n_blk = blockIdx.y * 64;

  auto issue = [&](int k0, int buf) {
    // A tile 128x32 halfs = 512 x 16B chunks -> 2 async instr / thread
#pragma unroll
    for (int it = 0; it < 2; ++it) {
      int i = tid + it * 256;
      int r = i >> 2, c = (i & 3) << 3;
      async_cp16(&sA[buf][r][c], A + (size_t)(m_blk + r) * K + k0 + c);
    }
    // B tile 64x32 halfs = 256 x 16B chunks -> 1 async instr / thread
    {
      int r = tid >> 2, c = (tid & 3) << 3;
      async_cp16(&sB[buf][r][c], Wt + (size_t)(n_blk + r) * K + k0 + c);
    }
  };

  v8f acc[2][2];
#pragma unroll
  for (int i = 0; i < 2; ++i)
#pragma unroll
    for (int j = 0; j < 2; ++j) acc[i][j] = (v8f){};

  const int nk = K >> 5;
  issue(0, 0);
  for (int kt = 0; kt < nk; ++kt) {
    int cur = kt & 1;
    if (kt + 1 < nk) {
      issue((kt + 1) << 5, cur ^ 1);
      asm volatile("s_wait_asynccnt 0x3" ::: "memory");   // prev tile (3 instr) retired
    } else {
      asm volatile("s_wait_asynccnt 0x0" ::: "memory");
    }
    __syncthreads();
    v16h af[2], bf[2];
#pragma unroll
    for (int i = 0; i < 2; ++i) af[i] = frag_a(&sA[cur][wm * 32 + i * 16 + l16][0], half);
#pragma unroll
    for (int j = 0; j < 2; ++j) bf[j] = frag_b(&sB[cur][wn * 32 + j * 16 + l16][0], half);
#pragma unroll
    for (int i = 0; i < 2; ++i)
#pragma unroll
      for (int j = 0; j < 2; ++j) acc[i][j] = wmma_f16(af[i], bf[j], acc[i][j]);
    __syncthreads();
  }
  // Epilogue: C-layout VGPR v -> row v+8*half, col lane&15
#pragma unroll
  for (int i = 0; i < 2; ++i)
#pragma unroll
    for (int j = 0; j < 2; ++j) {
      int gn = n_blk + wn * 32 + j * 16 + l16;
      float bv = bias ? bias[gn] : 0.f;
#pragma unroll
      for (int v = 0; v < 8; ++v) {
        int gm = m_blk + wm * 32 + i * 16 + v + 8 * half;
        float r = acc[i][j][v] + bv;
        if (relu) r = fmaxf(r, 0.f);
        if (Cf) Cf[(size_t)gm * N + gn] = r;
        if (Ch) Ch[(size_t)gm * N + gn] = (_Float16)r;
      }
    }
}

// ---------------------------------------------------------------------------
// Fused attention, one workgroup per (batch, head): QK^T -> softmax -> AV in LDS.
// Q,K,V f16 [B,L,256] (head h = cols h*32..+31); O f16. Q/K rows staged with
// async LDS copies (4 x b128 per row).
// ---------------------------------------------------------------------------
#define AQ_MAX 208
#define AKP_MAX 224
__global__ __launch_bounds__(256)
void attn_kernel(const _Float16* __restrict__ Q, const _Float16* __restrict__ Km,
                 const _Float16* __restrict__ V, _Float16* __restrict__ O,
                 const int* __restrict__ mask, int H, int Lq, int Lk,
                 int maskStride, float scale)
{
  __shared__ __align__(16) _Float16 sQ[AQ_MAX][32];
  __shared__ __align__(16) _Float16 sK[AQ_MAX][32];
  __shared__ __align__(16) _Float16 sVt[32][AKP_MAX];
  __shared__ __align__(16) _Float16 sS[AQ_MAX][AKP_MAX];

  const int tid = threadIdx.x;
  const int wave = tid >> 5, lane = tid & 31;
  const int half = lane >> 4, l16 = lane & 15;
  const int b = blockIdx.x / H, h = blockIdx.x % H;
  const int TQ = (Lq + 15) >> 4, TK = (Lk + 15) >> 4;
  const int KP = (Lk + 31) & ~31;
  const size_t E = 256;
  const _Float16* Qb = Q + ((size_t)b * Lq) * E + h * 32;
  const _Float16* Kb = Km + ((size_t)b * Lk) * E + h * 32;
  const _Float16* Vb = V + ((size_t)b * Lk) * E + h * 32;

  for (int i = tid; i < (TQ << 4) * 4; i += 256) {      // 4 x 16B chunks per row
    int q = i >> 2, c = (i & 3) << 3;
    if (q < Lq) async_cp16(&sQ[q][c], Qb + (size_t)q * E + c);
    else        { v8h z = {}; *(v8h*)&sQ[q][c] = z; }
  }
  for (int i = tid; i < (TK << 4) * 4; i += 256) {
    int k = i >> 2, c = (i & 3) << 3;
    if (k < Lk) async_cp16(&sK[k][c], Kb + (size_t)k * E + c);
    else        { v8h z = {}; *(v8h*)&sK[k][c] = z; }
  }
  for (int i = tid; i < KP * 32; i += 256) {            // V transposed manually
    int k = i >> 5, d = i & 31;
    sVt[d][k] = (k < Lk) ? Vb[(size_t)k * E + d] : (_Float16)0.f;
  }
  asm volatile("s_wait_asynccnt 0x0" ::: "memory");
  __syncthreads();

  // S = scale * Q K^T  (one WMMA per 16x16 tile, K-dim = 32 = head dim)
  const int ntile = TQ * TK;
  for (int t = wave; t < ntile; t += 8) {
    int ti = t / TK, tj = t % TK;
    v16h a  = frag_a(&sQ[ti * 16 + l16][0], half);
    v16h bf = frag_b(&sK[tj * 16 + l16][0], half);
    v8f c = (v8f){};
    c = wmma_f16(a, bf, c);
    int col = tj * 16 + l16;
    bool dead = (col >= Lk);
    if (!dead && mask) dead = (mask[(size_t)b * maskStride + col] == 0);
#pragma unroll
    for (int v = 0; v < 8; ++v) {
      float s = c[v] * scale;
      if (dead) s = -30000.f;                 // finite in f16; exp -> 0
      sS[ti * 16 + v + 8 * half][col] = (_Float16)s;
    }
  }
  __syncthreads();

  // row softmax (thread per row); zero padded columns for the AV WMMA
  for (int r = tid; r < (TQ << 4); r += 256) {
    if (r >= Lq) {
      for (int j = 0; j < KP; ++j) sS[r][j] = (_Float16)0.f;
    } else {
      float mx = -1e30f;
      for (int j = 0; j < Lk; ++j) mx = fmaxf(mx, (float)sS[r][j]);
      float sum = 0.f;
      for (int j = 0; j < Lk; ++j) {
        float e = __expf((float)sS[r][j] - mx);
        sum += e; sS[r][j] = (_Float16)e;
      }
      float inv = 1.f / sum;
      for (int j = 0; j < Lk; ++j) sS[r][j] = (_Float16)((float)sS[r][j] * inv);
      for (int j = Lk; j < KP; ++j) sS[r][j] = (_Float16)0.f;
    }
  }
  __syncthreads();

  // O = P @ V
  const int ntask = TQ * 2;
  for (int t = wave; t < ntask; t += 8) {
    int ti = t >> 1, tj = t & 1;
    v8f c = (v8f){};
    for (int ks = 0; ks < KP; ks += 32) {
      v16h a  = frag_a(&sS[ti * 16 + l16][ks], half);
      v16h bf = frag_b(&sVt[tj * 16 + l16][ks], half);
      c = wmma_f16(a, bf, c);
    }
    int d = tj * 16 + l16;
#pragma unroll
    for (int v = 0; v < 8; ++v) {
      int q = ti * 16 + v + 8 * half;
      if (q < Lq) O[((size_t)b * Lq + q) * E + h * 32 + d] = (_Float16)c[v];
    }
  }
}

// ---------------------------------------------------------------------------
// x = LayerNorm(x + r) * g + b  (wave per token, E=256); also writes f16 copy.
// ---------------------------------------------------------------------------
__global__ __launch_bounds__(256)
void add_ln_kernel(float* __restrict__ x, _Float16* __restrict__ xh,
                   const float* __restrict__ r,
                   const float* __restrict__ g, const float* __restrict__ bb, int T)
{
  int token = (blockIdx.x * 256 + threadIdx.x) >> 5;
  int lane = threadIdx.x & 31;
  if (token >= T) return;
  float* xp = x + (size_t)token * 256;
  const float* rp = r + (size_t)token * 256;
  _Float16* hp = xh + (size_t)token * 256;
  float v[8]; float s = 0.f, s2 = 0.f;
#pragma unroll
  for (int e = 0; e < 8; ++e) {
    int c = e * 32 + lane;
    float t = xp[c] + rp[c];
    v[e] = t; s += t; s2 += t * t;
  }
#pragma unroll
  for (int o = 16; o > 0; o >>= 1) { s += __shfl_xor(s, o, 32); s2 += __shfl_xor(s2, o, 32); }
  float mu = s * (1.f / 256.f);
  float var = s2 * (1.f / 256.f) - mu * mu;
  float inv = rsqrtf(var + 1e-5f);
#pragma unroll
  for (int e = 0; e < 8; ++e) {
    int c = e * 32 + lane;
    float o = (v[e] - mu) * inv * g[c] + bb[c];
    xp[c] = o;
    hp[c] = (_Float16)o;
  }
}

// out[row] = X[row,:256] . w + bias[0]   (wave per row)
__global__ __launch_bounds__(256)
void rowdot_kernel(const float* __restrict__ X, const float* __restrict__ w,
                   const float* __restrict__ bias, float* __restrict__ out, int T)
{
  int row = (blockIdx.x * 256 + threadIdx.x) >> 5;
  int lane = threadIdx.x & 31;
  if (row >= T) return;
  const float* xp = X + (size_t)row * 256;
  float s = 0.f;
#pragma unroll
  for (int e = 0; e < 8; ++e) s += xp[e * 32 + lane] * w[e * 32 + lane];
#pragma unroll
  for (int o = 16; o > 0; o >>= 1) s += __shfl_xor(s, o, 32);
  if (lane == 0) out[row] = s + bias[0];
}

// per-batch softmax over L entries (1 wave per batch)
__global__ void softmax_rows_kernel(const float* __restrict__ in, float* __restrict__ out, int L)
{
  int b = blockIdx.x, lane = threadIdx.x;
  float mx = -1e30f;
  for (int j = lane; j < L; j += 32) mx = fmaxf(mx, in[(size_t)b * L + j]);
#pragma unroll
  for (int o = 16; o > 0; o >>= 1) mx = fmaxf(mx, __shfl_xor(mx, o, 32));
  float s = 0.f;
  for (int j = lane; j < L; j += 32) s += __expf(in[(size_t)b * L + j] - mx);
#pragma unroll
  for (int o = 16; o > 0; o >>= 1) s += __shfl_xor(s, o, 32);
  float inv = 1.f / s;
  for (int j = lane; j < L; j += 32)
    out[(size_t)b * L + j] = __expf(in[(size_t)b * L + j] - mx) * inv;
}

// top-k selection (lax.top_k semantics: descending, lower index wins ties)
__global__ __launch_bounds__(256)
void topk_kernel(const float* __restrict__ scores, int L, int Kk, int* __restrict__ idx_out)
{
  __shared__ float sv[256];
  __shared__ float rv[256];
  __shared__ int   ri[256];
  int b = blockIdx.x, t = threadIdx.x;
  sv[t] = (t < L) ? scores[(size_t)b * L + t] : -3.0e38f;
  __syncthreads();
  for (int r = 0; r < Kk; ++r) {
    rv[t] = sv[t]; ri[t] = t;
    __syncthreads();
    for (int s = 128; s > 0; s >>= 1) {
      if (t < s) {
        float v2 = rv[t + s]; int i2 = ri[t + s];
        if (v2 > rv[t] || (v2 == rv[t] && i2 < ri[t])) { rv[t] = v2; ri[t] = i2; }
      }
      __syncthreads();
    }
    if (t == 0) { idx_out[(size_t)b * Kk + r] = ri[0]; sv[ri[0]] = -3.0e38f; }
    __syncthreads();
  }
}

// Y[b*Kk+r,:] = X[b*Lin+idx,:] (fp32, optional f16 copy); 64 threads, float4
__global__ void gather_kernel(const float* __restrict__ X, const int* __restrict__ idx,
                              float* __restrict__ Y, _Float16* __restrict__ Yh,
                              int Lin, int Kk)
{
  int row = blockIdx.x;
  int b = row / Kk;
  int src = idx[row];
  const float4* s = (const float4*)(X + ((size_t)b * Lin + src) * 256);
  float4 v = s[threadIdx.x];
  ((float4*)(Y + (size_t)row * 256))[threadIdx.x] = v;
  if (Yh) {
    _Float16* d = Yh + (size_t)row * 256 + threadIdx.x * 4;
    d[0] = (_Float16)v.x; d[1] = (_Float16)v.y; d[2] = (_Float16)v.z; d[3] = (_Float16)v.w;
  }
}

// padded embed input (f16) [B*M, 64]: [ped(16) | mode(24) | zeros(24)]
__global__ __launch_bounds__(256)
void build_pad_kernel(const float* __restrict__ ped, const float* __restrict__ modes,
                      _Float16* __restrict__ out)
{
  int i = blockIdx.x * 256 + threadIdx.x;            // < 51200*64
  int token = i >> 6, c = i & 63;
  int b = token / 200, m = token % 200;
  float v = 0.f;
  if (c < 16)      v = ped[(size_t)b * 16 + c];
  else if (c < 40) v = modes[(size_t)m * 24 + (c - 16)];
  out[i] = (_Float16)v;
}

// We [40,256] -> transposed+padded f16 [256,64]
__global__ void build_wept_kernel(const float* __restrict__ We, _Float16* __restrict__ Wt)
{
  int i = blockIdx.x * 256 + threadIdx.x;            // < 256*64
  int n = i >> 6, k = i & 63;
  Wt[i] = (k < 40) ? (_Float16)We[(size_t)k * 256 + n] : (_Float16)0.f;
}

// W [K,N] fp32 -> Wt [N,K] f16 (one-time weight conversion)
__global__ __launch_bounds__(256)
void wt_kernel(const float* __restrict__ W, _Float16* __restrict__ Wt, int K, int N)
{
  int i = blockIdx.x * 256 + threadIdx.x;
  if (i >= K * N) return;
  int k = i / N, n = i % N;
  Wt[(size_t)n * K + k] = (_Float16)W[i];
}

// nei (f16) = neis[B*50,16] @ Wn[16,256] + bn
__global__ __launch_bounds__(256)
void nei_embed_kernel(const float* __restrict__ neis, const float* __restrict__ Wn,
                      const float* __restrict__ bn, _Float16* __restrict__ out, int total)
{
  int i = blockIdx.x * 256 + threadIdx.x;
  if (i >= total) return;
  int tok = i >> 8, c = i & 255;
  const float* x = neis + (size_t)tok * 16;
  float s = bn[c];
#pragma unroll
  for (int k = 0; k < 16; ++k) s += x[k] * Wn[k * 256 + c];
  out[i] = (_Float16)s;
}

// pred = TOP2[B*20,256] @ Wr[256,24] + br
__global__ __launch_bounds__(256)
void pred_kernel(const float* __restrict__ T2, const float* __restrict__ Wr,
                 const float* __restrict__ br, float* __restrict__ out, int total)
{
  int i = blockIdx.x * 256 + threadIdx.x;
  if (i >= total) return;
  int row = i / 24, c = i % 24;
  const float* x = T2 + (size_t)row * 256;
  float s = 0.f;
  for (int k = 0; k < 256; ++k) s += x[k] * Wr[k * 24 + c];
  out[i] = s + br[c];
}

// ---------------------------------------------------------------------------
// Host orchestration
// ---------------------------------------------------------------------------
struct LayerP { const float *W1,*W2,*Wk,*Wo,*Wq,*Wv,*b1,*b2,*g1,*g2,*n1,*n2; };
struct LayerW { _Float16 *Wqt,*Wkt,*Wvt,*Wot,*W1t,*W2t; };

extern "C" void kernel_launch(void* const* d_in, const int* in_sizes, int n_in,
                              void* d_out, int out_size, void* d_ws, size_t ws_size,
                              hipStream_t stream)
{
  (void)in_sizes; (void)n_in; (void)out_size; (void)ws_size;
  const float* ped   = (const float*)d_in[0];
  const float* neis  = (const float*)d_in[1];
  const float* modes = (const float*)d_in[2];
  const int*   maskp = (const int*)d_in[3];
  const float* Wc  = (const float*)d_in[7];
  const float* Wc2 = (const float*)d_in[8];
  const float* We  = (const float*)d_in[9];
  const float* Wn  = (const float*)d_in[10];
  const float* Wr  = (const float*)d_in[11];
  const float* bc  = (const float*)d_in[12];
  const float* bc2 = (const float*)d_in[13];
  const float* be  = (const float*)d_in[14];
  const float* bn  = (const float*)d_in[15];
  const float* br  = (const float*)d_in[16];
  auto getL = [&](int base) {
    LayerP L;
    L.W1 = (const float*)d_in[base + 0]; L.W2 = (const float*)d_in[base + 1];
    L.Wk = (const float*)d_in[base + 2]; L.Wo = (const float*)d_in[base + 3];
    L.Wq = (const float*)d_in[base + 4]; L.Wv = (const float*)d_in[base + 5];
    L.b1 = (const float*)d_in[base + 6]; L.b2 = (const float*)d_in[base + 7];
    L.g1 = (const float*)d_in[base + 8]; L.g2 = (const float*)d_in[base + 9];
    L.n1 = (const float*)d_in[base +10]; L.n2 = (const float*)d_in[base +11];
    return L;
  };
  LayerP dec0 = getL(17), dec1 = getL(29), enc0 = getL(41), enc1 = getL(53);

  // --- workspace (byte offsets, 256B aligned)
  char* ws = (char*)d_ws;
  size_t off = 0;
  auto alloc = [&](size_t bytes) { char* p = ws + off; off += (bytes + 255) & ~(size_t)255; return p; };
  float*    X    = (float*)   alloc((size_t)51200 * 256 * 4);
  _Float16* Xh   = (_Float16*)alloc((size_t)51200 * 256 * 2);
  _Float16* Qh   = (_Float16*)alloc((size_t)51200 * 256 * 2);
  _Float16* Kh   = (_Float16*)alloc((size_t)51200 * 256 * 2);
  _Float16* Vh   = (_Float16*)alloc((size_t)51200 * 256 * 2);
  _Float16* Oh   = (_Float16*)alloc((size_t)51200 * 256 * 2);
  _Float16* Hh   = (_Float16*)alloc((size_t)12800 * 1024 * 2);
  float*    Tf   = (float*)   alloc((size_t)51200 * 256 * 4);
  _Float16* NEIh = (_Float16*)alloc((size_t)12800 * 256 * 2);
  float*    Y    = (float*)   alloc((size_t)25600 * 256 * 4);
  _Float16* Yh   = (_Float16*)alloc((size_t)25600 * 256 * 2);
  _Float16* PADh = (_Float16*)alloc((size_t)51200 * 64 * 2);
  _Float16* WEPt = (_Float16*)alloc((size_t)256 * 64 * 2);
  float*    SC   = (float*)   alloc((size_t)51200 * 4);
  float*    S2   = (float*)   alloc((size_t)25600 * 4);
  float*    T2   = (float*)   alloc((size_t)5120 * 256 * 4);
  int*      IDX1 = (int*)     alloc((size_t)25600 * 4);
  int*      IDX2 = (int*)     alloc((size_t)5120 * 4);
  LayerW lw[4];
  for (int l = 0; l < 4; ++l) {
    lw[l].Wqt = (_Float16*)alloc((size_t)256 * 256 * 2);
    lw[l].Wkt = (_Float16*)alloc((size_t)256 * 256 * 2);
    lw[l].Wvt = (_Float16*)alloc((size_t)256 * 256 * 2);
    lw[l].Wot = (_Float16*)alloc((size_t)256 * 256 * 2);
    lw[l].W1t = (_Float16*)alloc((size_t)1024 * 256 * 2);  // [1024n][256k]
    lw[l].W2t = (_Float16*)alloc((size_t)256 * 1024 * 2);  // [256n][1024k]
  }

  const float scale = 0.17677669529663687f;   // 1/sqrt(32)

  auto conv = [&](const float* W, _Float16* Wt, int K, int N) {
    wt_kernel<<<(K * N + 255) / 256, 256, 0, stream>>>(W, Wt, K, N);
  };
  LayerP lp[4] = { dec0, dec1, enc0, enc1 };
  for (int l = 0; l < 4; ++l) {
    conv(lp[l].Wq, lw[l].Wqt, 256, 256);
    conv(lp[l].Wk, lw[l].Wkt, 256, 256);
    conv(lp[l].Wv, lw[l].Wvt, 256, 256);
    conv(lp[l].Wo, lw[l].Wot, 256, 256);
    conv(lp[l].W1, lw[l].W1t, 256, 1024);
    conv(lp[l].W2, lw[l].W2t, 1024, 256);
  }
  build_wept_kernel<<<(256 * 64) / 256, 256, 0, stream>>>(We, WEPt);
  build_pad_kernel<<<(51200 * 64) / 256, 256, 0, stream>>>(ped, modes, PADh);

  auto gemm = [&](const _Float16* A, const _Float16* Wt, const float* bias,
                  float* Cf, _Float16* Ch, int M, int N, int K, int relu) {
    gemm_kernel<<<dim3(M / 128, N / 64), 256, 0, stream>>>(A, Wt, bias, Cf, Ch, M, N, K, relu);
  };
  auto block = [&](float* Xf, _Float16* Xfh, const _Float16* KVh, const LayerP& L,
                   const LayerW& W, int B, int Lq, int Lk, const int* mk, int mstride) {
    int Tq = B * Lq, Tk = B * Lk;
    gemm(Xfh, W.Wqt, nullptr, nullptr, Qh, Tq, 256, 256, 0);
    gemm(KVh, W.Wkt, nullptr, nullptr, Kh, Tk, 256, 256, 0);
    gemm(KVh, W.Wvt, nullptr, nullptr, Vh, Tk, 256, 256, 0);
    attn_kernel<<<B * 8, 256, 0, stream>>>(Qh, Kh, Vh, Oh, mk, 8, Lq, Lk, mstride, scale);
    gemm(Oh, W.Wot, nullptr, Tf, nullptr, Tq, 256, 256, 0);
    add_ln_kernel<<<(Tq + 7) / 8, 256, 0, stream>>>(Xf, Xfh, Tf, L.g1, L.n1, Tq);
    for (int c0 = 0; c0 < Tq; c0 += 12800) {                      // FFN, token-chunked
      gemm(Xfh + (size_t)c0 * 256, W.W1t, L.b1, nullptr, Hh, 12800, 1024, 256, 1);
      gemm(Hh, W.W2t, L.b2, Tf + (size_t)c0 * 256, nullptr, 12800, 256, 1024, 0);
    }
    add_ln_kernel<<<(Tq + 7) / 8, 256, 0, stream>>>(Xf, Xfh, Tf, L.g2, L.n2, Tq);
  };

  // 1. embedding: X = concat(ped, modes) @ We + be  (K padded 40->64)
  gemm(PADh, WEPt, be, X, Xh, 51200, 256, 64, 0);

  // 2. encoder (self-attn over 200 modes, B=256)
  block(X, Xh, Xh, enc0, lw[2], 256, 200, 200, nullptr, 0);
  block(X, Xh, Xh, enc1, lw[3], 256, 200, 200, nullptr, 0);

  // 3. scores -> top-100 -> gather
  rowdot_kernel<<<51200 / 8, 256, 0, stream>>>(X, Wc, bc, SC, 51200);
  topk_kernel<<<256, 256, 0, stream>>>(SC, 200, 100, IDX1);
  gather_kernel<<<256 * 100, 64, 0, stream>>>(X, IDX1, Y, Yh, 200, 100);

  // 4. neighbor embedding (f16)
  nei_embed_kernel<<<(256 * 50 * 256) / 256, 256, 0, stream>>>(neis, Wn, bn, NEIh, 256 * 50 * 256);

  // 5. decoder (cross-attn: 100 queries x 50 masked keys)
  block(Y, Yh, NEIh, dec0, lw[0], 256, 100, 50, maskp, 2500);
  block(Y, Yh, NEIh, dec1, lw[1], 256, 100, 50, maskp, 2500);

  // 6. scores_nei = softmax(Y @ Wc2 + bc2)  -> second output region
  float* out_scores = (float*)d_out + 256 * 20 * 24;
  rowdot_kernel<<<25600 / 8, 256, 0, stream>>>(Y, Wc2, bc2, S2, 25600);
  softmax_rows_kernel<<<256, 32, 0, stream>>>(S2, out_scores, 100);

  // 7. top-20 -> gather -> prediction head -> first output region
  topk_kernel<<<256, 256, 0, stream>>>(out_scores, 100, 20, IDX2);
  gather_kernel<<<256 * 20, 64, 0, stream>>>(Y, IDX2, T2, nullptr, 100, 20);
  pred_kernel<<<(256 * 20 * 24 + 255) / 256, 256, 0, stream>>>(T2, Wr, br, (float*)d_out, 256 * 20 * 24);
}
